// CLIPMaskedSpatialViT_13649406066871
// MI455X (gfx1250) — matrix operand order repaired and checked by reading, not compiled, not run on hardware
//
#include <hip/hip_runtime.h>

typedef __attribute__((ext_vector_type(16))) _Float16 v16h;
typedef __attribute__((ext_vector_type(8)))  _Float16 v8h;
typedef __attribute__((ext_vector_type(8)))  float    v8f;

#define CDIV(a, b) (((a) + (b) - 1) / (b))

// Model constants
#define NLAYER 12
#define DMODEL 768
#define FFDIM  3072
#define NHEAD  12
#define HDIM   64
#define BATCH  8
#define NTOK   197                 // cls + 196 patches
#define TROWS  (NTOK * BATCH)      // 1576
#define TROWSP 1664                // ceil(1576/128)*128
#define KPAD   256                 // padded token/key count (mult of 128)
#define NZ     (BATCH * NHEAD)     // 96 batched attention problems
#define SRCT   212                 // 196 + 16 source tokens for masked branch
#define SRCROWS (SRCT * BATCH)     // 1696
#define SRCROWSP 1792              // ceil(1696/128)*128
#define NMASK  16
#define GVROWS (NMASK * BATCH)     // 128
#define GQROWS 128                 // row pitch for tiny gv-branch A operands
#define OUTD   512
#define CONVROWS (BATCH * 196)     // 1568
#define CONVROWSP 1664

// ===========================================================================
// WMMA GEMM:  C[z] = alpha * A[z] @ B[z]^T (+bias[n]) (+QuickGELU)
//   A: [Mpad,K] f16 row-major; rows in [M, ceil(M/128)*128) MUST be zero.
//   B: [N,K]    f16 row-major (computes A @ B^T); N multiple of 64.
//   C: [M,N]    f32 (or f16 when out16) row-major; only rows < M stored.
//   K multiple of 32.
// Block = 128 threads (4 waves). Each wave owns a 32(M) x 64(N) strip
// (two 16-row tiles x four 16-col tiles = 8 accumulators), so each B
// fragment fetched feeds two WMMAs. Block covers 128 M rows x 64 N cols.
// No control flow inside the k-loop.
// ===========================================================================
__global__ __launch_bounds__(128) void wmma_gemm_kernel(
    const _Float16* __restrict__ A, const _Float16* __restrict__ B,
    const float* __restrict__ bias, void* __restrict__ Cout,
    int M, int N, int K, long sA, long sB, long sC,
    float alpha, int act, int out16)
{
  const int z    = blockIdx.z;
  const int wave = threadIdx.x >> 5;
  const int lane = threadIdx.x & 31;
  const int lm   = lane & 15;
  const int lhi  = lane >> 4;
  const int mTile = blockIdx.y * 128 + wave * 32;   // two 16-row tiles
  const int nBase = blockIdx.x * 64;

  // A fragment (16-bit A 16x32): lanes 0-15 rows m, K=k0..k0+15;
  // lanes 16-31 same rows, K=k0+16..k0+31 (matches B).
  const _Float16* ap0 = A + (size_t)z * sA + (size_t)(mTile + lm) * K + (lhi << 4);
  const _Float16* ap1 = ap0 + (size_t)16 * K;
  // B fragment (16-bit B 32x16): lane n = lane&15; lanes 0-15 K=k0..k0+15,
  // lanes 16-31 K=k0+16..k0+31 -> 16 contiguous halves of row n.
  const _Float16* bp = B + (size_t)z * sB + (size_t)(nBase + lm) * K + (lhi << 4);
  const size_t bStep = (size_t)16 * K;

  v8f acc[8] = {};

  for (int k0 = 0; k0 < K; k0 += 32) {
    v8h a0l = *(const v8h*)(ap0);
    v8h a0h = *(const v8h*)(ap0 + 8);
    v8h a1l = *(const v8h*)(ap1);
    v8h a1h = *(const v8h*)(ap1 + 8);
    v16h a0 = __builtin_shufflevector(a0l, a0h, 0,1,2,3,4,5,6,7,8,9,10,11,12,13,14,15);
    v16h a1 = __builtin_shufflevector(a1l, a1h, 0,1,2,3,4,5,6,7,8,9,10,11,12,13,14,15);
    v16h b0 = *(const v16h*)(bp);
    v16h b1 = *(const v16h*)(bp + bStep);
    v16h b2 = *(const v16h*)(bp + 2 * bStep);
    v16h b3 = *(const v16h*)(bp + 3 * bStep);
    acc[0] = __builtin_amdgcn_wmma_f32_16x16x32_f16(false, a0, false, b0, (short)0, acc[0], false, false);
    acc[4] = __builtin_amdgcn_wmma_f32_16x16x32_f16(false, a1, false, b0, (short)0, acc[4], false, false);
    acc[1] = __builtin_amdgcn_wmma_f32_16x16x32_f16(false, a0, false, b1, (short)0, acc[1], false, false);
    acc[5] = __builtin_amdgcn_wmma_f32_16x16x32_f16(false, a1, false, b1, (short)0, acc[5], false, false);
    acc[2] = __builtin_amdgcn_wmma_f32_16x16x32_f16(false, a0, false, b2, (short)0, acc[2], false, false);
    acc[6] = __builtin_amdgcn_wmma_f32_16x16x32_f16(false, a1, false, b2, (short)0, acc[6], false, false);
    acc[3] = __builtin_amdgcn_wmma_f32_16x16x32_f16(false, a0, false, b3, (short)0, acc[3], false, false);
    acc[7] = __builtin_amdgcn_wmma_f32_16x16x32_f16(false, a1, false, b3, (short)0, acc[7], false, false);
    ap0 += 32;
    ap1 += 32;
    bp  += 32;
  }

  // store: f32 C/D 16x16 layout: VGPR r -> row tileBase + 8*(lane>>4) + r, col = lane&15
#pragma unroll
  for (int t = 0; t < 2; ++t) {
    const int mBase = mTile + t * 16 + (lhi << 3);
#pragma unroll
    for (int j = 0; j < 4; ++j) {
      const int n = nBase + j * 16 + lm;
      const float bv = bias ? bias[n] : 0.0f;
      const v8f a = acc[t * 4 + j];
#pragma unroll
      for (int r = 0; r < 8; ++r) {
        const int mm = mBase + r;
        if (mm < M) {
          float v = a[r] * alpha + bv;
          if (act) v = v * (1.0f / (1.0f + __expf(-1.702f * v)));  // QuickGELU
          const size_t idx = (size_t)z * sC + (size_t)mm * N + n;
          if (out16) ((_Float16*)Cout)[idx] = (_Float16)v;
          else       ((float*)Cout)[idx] = v;
        }
      }
    }
  }
}

// ===========================================================================
// Elementwise / reduction helper kernels
// ===========================================================================
__global__ void cvt_f32_f16_kernel(const float* __restrict__ s,
                                   _Float16* __restrict__ d, size_t n) {
  size_t i = (size_t)blockIdx.x * blockDim.x + threadIdx.x;
  if (i < n) d[i] = (_Float16)s[i];
}

// s [R][C] f32 -> d [C][R] f16  (proj: [768,512] -> [512,768])
__global__ void transpose_f16_kernel(const float* __restrict__ s,
                                     _Float16* __restrict__ d, int R, int C) {
  size_t i = (size_t)blockIdx.x * blockDim.x + threadIdx.x;
  if (i >= (size_t)R * C) return;
  int r = (int)(i / C), c = (int)(i % C);
  d[(size_t)c * R + r] = (_Float16)s[i];
}

__global__ void fill_f16_kernel(_Float16* p, size_t n) {
  size_t i = (size_t)blockIdx.x * blockDim.x + threadIdx.x;
  if (i < n) p[i] = (_Float16)0.0f;
}

// im2col for conv: stride 16, kernel 16, pad 7.  out f16 [CONVROWSP][768]
// (pad rows pre-zeroed), k = c*256 + kh*16 + kw
__global__ void im2col_kernel(const float* __restrict__ im, _Float16* __restrict__ out) {
  size_t i = (size_t)blockIdx.x * blockDim.x + threadIdx.x;
  const size_t total = (size_t)CONVROWS * DMODEL;
  if (i >= total) return;
  int k = (int)(i % DMODEL);
  int row = (int)(i / DMODEL);
  int b = row / 196, p = row % 196;
  int c = k >> 8, kh = (k >> 4) & 15, kw = k & 15;
  int iy = (p / 14) * 16 - 7 + kh;
  int ix = (p % 14) * 16 - 7 + kw;
  float v = 0.0f;
  if (iy >= 0 && iy < 224 && ix >= 0 && ix < 224)
    v = im[(((size_t)b * 3 + c) * 224 + iy) * 224 + ix];
  out[i] = (_Float16)v;
}

// pre[t*B+b][d] = (t==0 ? cls[d] : conv[(b*196 + t-1)][d]) + pos[t][d]
__global__ void build_x_pre_kernel(const float* __restrict__ conv,
                                   const float* __restrict__ cls,
                                   const float* __restrict__ pos,
                                   float* __restrict__ pre) {
  size_t i = (size_t)blockIdx.x * blockDim.x + threadIdx.x;
  const size_t total = (size_t)TROWS * DMODEL;
  if (i >= total) return;
  int d = (int)(i % DMODEL);
  int row = (int)(i / DMODEL);
  int t = row / BATCH, b = row % BATCH;
  float v = (t == 0) ? cls[d] : conv[((size_t)b * 196 + (t - 1)) * DMODEL + d];
  pre[i] = v + pos[(size_t)t * DMODEL + d];
}

// One block per row. Rows >= Rvalid write zeros (A-pad rows for the GEMM).
// Writes y16 (f16, for WMMA A) and/or y32 (f32 residual stream) if non-null.
__global__ __launch_bounds__(256) void layernorm_kernel(
    const float* __restrict__ X, const float* __restrict__ g,
    const float* __restrict__ b, _Float16* __restrict__ y16,
    float* __restrict__ y32, int W, int Rvalid) {
  const int row = blockIdx.x;
  const int tid = threadIdx.x;
  if (row >= Rvalid) {
    if (y16) for (int j = tid; j < W; j += 256) y16[(size_t)row * W + j] = (_Float16)0.0f;
    if (y32) for (int j = tid; j < W; j += 256) y32[(size_t)row * W + j] = 0.0f;
    return;
  }
  const float* x = X + (size_t)row * W;
  __shared__ float red[256];
  float s = 0.0f;
  for (int j = tid; j < W; j += 256) s += x[j];
  red[tid] = s; __syncthreads();
  for (int st = 128; st > 0; st >>= 1) { if (tid < st) red[tid] += red[tid + st]; __syncthreads(); }
  float mean = red[0] / W;
  __syncthreads();
  float v = 0.0f;
  for (int j = tid; j < W; j += 256) { float d0 = x[j] - mean; v += d0 * d0; }
  red[tid] = v; __syncthreads();
  for (int st = 128; st > 0; st >>= 1) { if (tid < st) red[tid] += red[tid + st]; __syncthreads(); }
  float rstd = rsqrtf(red[0] / W + 1e-5f);
  for (int j = tid; j < W; j += 256) {
    float o = (x[j] - mean) * rstd * g[j] + b[j];
    if (y16) y16[(size_t)row * W + j] = (_Float16)o;
    if (y32) y32[(size_t)row * W + j] = o;
  }
}

// gv[t*B+b][d] = x[0*B+b][d]   (tile token 0)
__global__ void tile_gv_kernel(const float* __restrict__ x, float* __restrict__ gv) {
  size_t i = (size_t)blockIdx.x * blockDim.x + threadIdx.x;
  const size_t total = (size_t)GVROWS * DMODEL;
  if (i >= total) return;
  gv[i] = x[i % ((size_t)BATCH * DMODEL)];
}

// src rows: t<196 -> x[(t+1)*B+b]; t>=196 -> gv[(t-196)*B+b]
__global__ void concat_src_kernel(const float* __restrict__ x,
                                  const float* __restrict__ gv,
                                  float* __restrict__ src) {
  size_t i = (size_t)blockIdx.x * blockDim.x + threadIdx.x;
  const size_t total = (size_t)SRCROWS * DMODEL;
  if (i >= total) return;
  int d = (int)(i % DMODEL);
  int row = (int)(i / DMODEL);
  int t = row / BATCH, b = row % BATCH;
  src[i] = (t < 196) ? x[((size_t)(t + 1) * BATCH + b) * DMODEL + d]
                     : gv[((size_t)(t - 196) * BATCH + b) * DMODEL + d];
}

// am[n][j] (ld KPAD): j<196 downsampled mask, j>=196 off-diagonal -> -100
__global__ void attn_mask_kernel(const int* __restrict__ masks, float* __restrict__ am) {
  int n = blockIdx.x, j = threadIdx.x;
  if (j >= SRCT) return;
  float a;
  if (j < 196) {
    int r = j / 14, c = j % 14;
    int pix = masks[((size_t)n * 224 + r * 16) * 224 + c * 16];
    a = (pix != 0) ? 0.0f : 1.0f;
  } else {
    a = ((j - 196) == n) ? 0.0f : 1.0f;
  }
  am[(size_t)n * KPAD + j] = (a != 0.0f) ? -100.0f : 0.0f;
}

// qkv f32 [rows][2304] (row = t*B+b) ->
//  q16 [z][TPq][64] f16 (t>=qOff), k16 [z][KPAD][64] f16, vt [z][64][KPAD] f16
// (pad rows/keys stay zero from the one-time launch fill)
__global__ void scatter_qkv_kernel(const float* __restrict__ qkv,
                                   _Float16* __restrict__ q16,
                                   _Float16* __restrict__ k16,
                                   _Float16* __restrict__ vt,
                                   int qOff, int TPq, size_t n) {
  size_t i = (size_t)blockIdx.x * blockDim.x + threadIdx.x;
  if (i >= n) return;
  int d = (int)(i % DMODEL);
  int row = (int)(i / DMODEL);
  int t = row / BATCH, b = row % BATCH;
  int h = d >> 6, hd = d & 63;
  int z = b * NHEAD + h;
  const float* s = qkv + (size_t)row * (3 * DMODEL);
  k16[((size_t)z * KPAD + t) * HDIM + hd] = (_Float16)s[DMODEL + d];
  vt[((size_t)z * HDIM + hd) * KPAD + t]  = (_Float16)s[2 * DMODEL + d];
  if (t >= qOff)
    q16[((size_t)z * TPq + (t - qOff)) * HDIM + hd] = (_Float16)s[d];
}

// softmax over keys; S [z][sRows][KPAD] f32 -> P [z][pRows][KPAD] f16
__global__ __launch_bounds__(256) void softmax_kernel(
    const float* __restrict__ S, _Float16* __restrict__ P,
    const float* __restrict__ mask, int Kvalid, int sRows, int pRows, int maskLd) {
  int q = blockIdx.x, z = blockIdx.y;
  const float* srow = S + ((size_t)z * sRows + q) * KPAD;
  _Float16* prow = P + ((size_t)z * pRows + q) * KPAD;
  __shared__ float red[256];
  int tid = threadIdx.x;
  float mx = -1e30f;
  for (int j = tid; j < Kvalid; j += 256) {
    float v = srow[j] + (mask ? mask[(size_t)q * maskLd + j] : 0.0f);
    mx = fmaxf(mx, v);
  }
  red[tid] = mx; __syncthreads();
  for (int st = 128; st > 0; st >>= 1) { if (tid < st) red[tid] = fmaxf(red[tid], red[tid + st]); __syncthreads(); }
  mx = red[0];
  __syncthreads();
  float sum = 0.0f;
  for (int j = tid; j < Kvalid; j += 256)
    sum += __expf(srow[j] + (mask ? mask[(size_t)q * maskLd + j] : 0.0f) - mx);
  red[tid] = sum; __syncthreads();
  for (int st = 128; st > 0; st >>= 1) { if (tid < st) red[tid] += red[tid + st]; __syncthreads(); }
  float inv = 1.0f / red[0];
  for (int j = tid; j < KPAD; j += 256) {
    float v = (j < Kvalid)
      ? __expf(srow[j] + (mask ? mask[(size_t)q * maskLd + j] : 0.0f) - mx) * inv
      : 0.0f;
    prow[j] = (_Float16)v;
  }
}

// attn [z][TPq][64] f32 -> o16[(t*B+b)][h*64+hd] f16
__global__ void reshape_attn_kernel(const float* __restrict__ attn,
                                    _Float16* __restrict__ o, int TPq, size_t n) {
  size_t i = (size_t)blockIdx.x * blockDim.x + threadIdx.x;
  if (i >= n) return;
  int d = (int)(i % DMODEL);
  int row = (int)(i / DMODEL);
  int t = row / BATCH, b = row % BATCH;
  int h = d >> 6, hd = d & 63;
  int z = b * NHEAD + h;
  o[i] = (_Float16)attn[((size_t)z * TPq + t) * HDIM + hd];
}

__global__ void residual_add_kernel(float* __restrict__ y,
                                    const float* __restrict__ c, size_t n) {
  size_t i = (size_t)blockIdx.x * blockDim.x + threadIdx.x;
  if (i < n) y[i] += c[i];
}

// C rows t*B+b [128][512] -> out[b][t][512]
__global__ void final_reorder_kernel(const float* __restrict__ s, float* __restrict__ o) {
  size_t i = (size_t)blockIdx.x * blockDim.x + threadIdx.x;
  const size_t total = (size_t)BATCH * NMASK * OUTD;
  if (i >= total) return;
  int c = (int)(i % OUTD);
  int row = (int)(i / OUTD);
  int b = row / NMASK, t = row % NMASK;
  o[i] = s[((size_t)t * BATCH + b) * OUTD + c];
}

// ===========================================================================
// Host-side launch helper
// ===========================================================================
static void gemm(const _Float16* A, const _Float16* B, const float* bias, void* C,
                 int M, int N, int K, int batch, long sA, long sB, long sC,
                 float alpha, int act, int out16, hipStream_t st) {
  dim3 g(CDIV(N, 64), CDIV(M, 128), batch);
  wmma_gemm_kernel<<<g, 128, 0, st>>>(A, B, bias, C, M, N, K, sA, sB, sC, alpha, act, out16);
}

extern "C" void kernel_launch(void* const* d_in, const int* in_sizes, int n_in,
                              void* d_out, int out_size, void* d_ws, size_t ws_size,
                              hipStream_t stream) {
  (void)in_sizes; (void)n_in; (void)out_size; (void)ws_size;
  const float* im        = (const float*)d_in[0];
  const int*   masks     = (const int*)  d_in[1];
  const float* conv_w    = (const float*)d_in[2];
  const float* class_emb = (const float*)d_in[3];
  const float* pos_emb   = (const float*)d_in[4];
  const float* ln_pre_g  = (const float*)d_in[5];
  const float* ln_pre_b  = (const float*)d_in[6];
  const float* ln1_g     = (const float*)d_in[7];
  const float* ln1_b     = (const float*)d_in[8];
  const float* ln2_g     = (const float*)d_in[9];
  const float* ln2_b     = (const float*)d_in[10];
  const float* qkv_w     = (const float*)d_in[11];
  const float* qkv_b     = (const float*)d_in[12];
  const float* out_w     = (const float*)d_in[13];
  const float* out_b     = (const float*)d_in[14];
  const float* fc_w      = (const float*)d_in[15];
  const float* fc_b      = (const float*)d_in[16];
  const float* cproj_w   = (const float*)d_in[17];
  const float* cproj_b   = (const float*)d_in[18];
  const float* lnp_g     = (const float*)d_in[19];
  const float* lnp_b     = (const float*)d_in[20];
  const float* proj      = (const float*)d_in[21];
  float* out = (float*)d_out;

  // ---- bump allocator over workspace ----
  char* wsp = (char*)d_ws;
  size_t off = 0;
  auto alloc = [&](size_t bytes) -> void* {
    void* p = wsp + off;
    off = (off + bytes + 255) & ~(size_t)255;
    return p;
  };

  // f16 weights ([N,K] row-major already, except proj which is transposed)
  _Float16* wq16  = (_Float16*)alloc((size_t)NLAYER * 3 * DMODEL * DMODEL * 2);
  _Float16* wo16  = (_Float16*)alloc((size_t)NLAYER * DMODEL * DMODEL * 2);
  _Float16* wf16  = (_Float16*)alloc((size_t)NLAYER * FFDIM * DMODEL * 2);
  _Float16* wc16  = (_Float16*)alloc((size_t)NLAYER * DMODEL * FFDIM * 2);
  _Float16* wcv16 = (_Float16*)alloc((size_t)DMODEL * DMODEL * 2);
  _Float16* wp16  = (_Float16*)alloc((size_t)OUTD * DMODEL * 2);
  // f16 GEMM A operands (padded rows, zero-filled once at launch start)
  _Float16* col16 = (_Float16*)alloc((size_t)CONVROWSP * DMODEL * 2);
  _Float16* hbuf16= (_Float16*)alloc((size_t)SRCROWSP * DMODEL * 2);
  _Float16* q16   = (_Float16*)alloc((size_t)NZ * KPAD * HDIM * 2);
  _Float16* k16   = (_Float16*)alloc((size_t)NZ * KPAD * HDIM * 2);
  _Float16* vt16  = (_Float16*)alloc((size_t)NZ * HDIM * KPAD * 2);
  _Float16* probs = (_Float16*)alloc((size_t)NZ * KPAD * KPAD * 2);
  _Float16* qg16  = (_Float16*)alloc((size_t)NZ * GQROWS * HDIM * 2);
  _Float16* kg16  = (_Float16*)alloc((size_t)NZ * KPAD * HDIM * 2);
  _Float16* vgt16 = (_Float16*)alloc((size_t)NZ * HDIM * KPAD * 2);
  _Float16* pg    = (_Float16*)alloc((size_t)NZ * GQROWS * KPAD * 2);
  _Float16* obuf16= (_Float16*)alloc((size_t)TROWSP * DMODEL * 2);
  _Float16* mlph16= (_Float16*)alloc((size_t)TROWSP * FFDIM * 2);
  // f32 buffers
  float* am     = (float*)alloc((size_t)NMASK * KPAD * 4);
  float* cvout  = (float*)alloc((size_t)CONVROWS * DMODEL * 4);
  float* x      = (float*)alloc((size_t)TROWS * DMODEL * 4);
  float* gv     = (float*)alloc((size_t)GVROWS * DMODEL * 4);
  float* qkvb   = (float*)alloc((size_t)SRCROWS * 3 * DMODEL * 4);
  float* scores = (float*)alloc((size_t)NZ * KPAD * KPAD * 4);
  float* attn   = (float*)alloc((size_t)NZ * KPAD * HDIM * 4);
  float* sg     = (float*)alloc((size_t)NZ * NMASK * KPAD * 4);
  float* ag     = (float*)alloc((size_t)NZ * NMASK * HDIM * 4);
  float* cbuf   = (float*)alloc((size_t)TROWS * DMODEL * 4);
  float* srcb   = (float*)alloc((size_t)SRCROWS * DMODEL * 4);  // also "pre" buffer
  float* fcbuf  = (float*)alloc((size_t)GVROWS * OUTD * 4);

  auto cvt = [&](const float* s, _Float16* d, size_t n) {
    cvt_f32_f16_kernel<<<CDIV(n, 256), 256, 0, stream>>>(s, d, n);
  };
  auto zero16 = [&](_Float16* p, size_t n) {
    fill_f16_kernel<<<CDIV(n, 256), 256, 0, stream>>>(p, n);
  };

  // ---- one-time zero fill of all padded f16 A/B operand buffers ----
  zero16(col16, (size_t)CONVROWSP * DMODEL);
  zero16(q16,   (size_t)NZ * KPAD * HDIM);
  zero16(k16,   (size_t)NZ * KPAD * HDIM);
  zero16(vt16,  (size_t)NZ * HDIM * KPAD);
  zero16(probs, (size_t)NZ * KPAD * KPAD);
  zero16(qg16,  (size_t)NZ * GQROWS * HDIM);
  zero16(kg16,  (size_t)NZ * KPAD * HDIM);
  zero16(vgt16, (size_t)NZ * HDIM * KPAD);
  zero16(pg,    (size_t)NZ * GQROWS * KPAD);
  zero16(obuf16,(size_t)TROWSP * DMODEL);
  zero16(mlph16,(size_t)TROWSP * FFDIM);

  // ---- weight conversion ----
  cvt(qkv_w,  wq16,  (size_t)NLAYER * 3 * DMODEL * DMODEL);
  cvt(out_w,  wo16,  (size_t)NLAYER * DMODEL * DMODEL);
  cvt(fc_w,   wf16,  (size_t)NLAYER * FFDIM * DMODEL);
  cvt(cproj_w,wc16,  (size_t)NLAYER * DMODEL * FFDIM);
  cvt(conv_w, wcv16, (size_t)DMODEL * DMODEL);
  transpose_f16_kernel<<<CDIV((size_t)DMODEL * OUTD, 256), 256, 0, stream>>>(proj, wp16, DMODEL, OUTD);

  // ---- attention mask ----
  attn_mask_kernel<<<NMASK, 224, 0, stream>>>(masks, am);

  // ---- patch embedding: im2col + WMMA GEMM, then cls/pos + ln_pre ----
  {
    size_t n = (size_t)CONVROWS * DMODEL;
    im2col_kernel<<<CDIV(n, 256), 256, 0, stream>>>(im, col16);
    gemm(col16, wcv16, nullptr, cvout, CONVROWS, DMODEL, DMODEL, 1, 0, 0, 0, 1.0f, 0, 0, stream);
    size_t nx = (size_t)TROWS * DMODEL;
    build_x_pre_kernel<<<CDIV(nx, 256), 256, 0, stream>>>(cvout, class_emb, pos_emb, srcb);
    layernorm_kernel<<<TROWS, 256, 0, stream>>>(srcb, ln_pre_g, ln_pre_b, nullptr, x, DMODEL, TROWS);
    tile_gv_kernel<<<CDIV((size_t)GVROWS * DMODEL, 256), 256, 0, stream>>>(x, gv);
  }

  for (int l = 0; l < NLAYER; ++l) {
    const _Float16* Wq = wq16 + (size_t)l * 3 * DMODEL * DMODEL;
    const _Float16* Wo = wo16 + (size_t)l * DMODEL * DMODEL;
    const _Float16* Wf = wf16 + (size_t)l * FFDIM * DMODEL;
    const _Float16* Wc = wc16 + (size_t)l * DMODEL * FFDIM;
    const float* bq = qkv_b + (size_t)l * 3 * DMODEL;
    const float* bo = out_b + (size_t)l * DMODEL;
    const float* bf = fc_b + (size_t)l * FFDIM;
    const float* bc = cproj_b + (size_t)l * DMODEL;
    const float* g1 = ln1_g + (size_t)l * DMODEL;
    const float* b1 = ln1_b + (size_t)l * DMODEL;
    const float* g2 = ln2_g + (size_t)l * DMODEL;
    const float* b2 = ln2_b + (size_t)l * DMODEL;

    // ============ masked cross-attention branch (uses pre-update x) =========
    concat_src_kernel<<<CDIV((size_t)SRCROWS * DMODEL, 256), 256, 0, stream>>>(x, gv, srcb);
    layernorm_kernel<<<SRCROWSP, 256, 0, stream>>>(srcb, g1, b1, hbuf16, nullptr, DMODEL, SRCROWS);
    gemm(hbuf16, Wq, bq, qkvb, SRCROWS, 3 * DMODEL, DMODEL, 1, 0, 0, 0, 1.0f, 0, 0, stream);
    {
      size_t n = (size_t)SRCROWS * DMODEL;
      scatter_qkv_kernel<<<CDIV(n, 256), 256, 0, stream>>>(qkvb, qg16, kg16, vgt16, 196, GQROWS, n);
    }
    gemm(qg16, kg16, nullptr, sg, NMASK, KPAD, HDIM, NZ,
         (long)GQROWS * HDIM, (long)KPAD * HDIM, (long)NMASK * KPAD, 0.125f, 0, 0, stream);
    softmax_kernel<<<dim3(NMASK, NZ), 256, 0, stream>>>(sg, pg, am, SRCT, NMASK, GQROWS, KPAD);
    gemm(pg, vgt16, nullptr, ag, NMASK, HDIM, KPAD, NZ,
         (long)GQROWS * KPAD, (long)HDIM * KPAD, (long)NMASK * HDIM, 1.0f, 0, 0, stream);
    {
      size_t n = (size_t)GVROWS * DMODEL;
      reshape_attn_kernel<<<CDIV(n, 256), 256, 0, stream>>>(ag, obuf16, NMASK, n);
      gemm(obuf16, Wo, bo, cbuf, GVROWS, DMODEL, DMODEL, 1, 0, 0, 0, 1.0f, 0, 0, stream);
      residual_add_kernel<<<CDIV(n, 256), 256, 0, stream>>>(gv, cbuf, n);
      layernorm_kernel<<<GVROWS, 256, 0, stream>>>(gv, g2, b2, hbuf16, nullptr, DMODEL, GVROWS);
      gemm(hbuf16, Wf, bf, mlph16, GVROWS, FFDIM, DMODEL, 1, 0, 0, 0, 1.0f, 1, 1, stream);
      gemm(mlph16, Wc, bc, cbuf, GVROWS, DMODEL, FFDIM, 1, 0, 0, 0, 1.0f, 0, 0, stream);
      residual_add_kernel<<<CDIV(n, 256), 256, 0, stream>>>(gv, cbuf, n);
    }

    // ============ standard CLIP resblock on x ===============================
    layernorm_kernel<<<TROWSP, 256, 0, stream>>>(x, g1, b1, hbuf16, nullptr, DMODEL, TROWS);
    gemm(hbuf16, Wq, bq, qkvb, TROWS, 3 * DMODEL, DMODEL, 1, 0, 0, 0, 1.0f, 0, 0, stream);
    {
      size_t n = (size_t)TROWS * DMODEL;
      scatter_qkv_kernel<<<CDIV(n, 256), 256, 0, stream>>>(qkvb, q16, k16, vt16, 0, KPAD, n);
    }
    gemm(q16, k16, nullptr, scores, NTOK, KPAD, HDIM, NZ,
         (long)KPAD * HDIM, (long)KPAD * HDIM, (long)KPAD * KPAD, 0.125f, 0, 0, stream);
    softmax_kernel<<<dim3(NTOK, NZ), 256, 0, stream>>>(scores, probs, nullptr, NTOK, KPAD, KPAD, 0);
    gemm(probs, vt16, nullptr, attn, NTOK, HDIM, KPAD, NZ,
         (long)KPAD * KPAD, (long)HDIM * KPAD, (long)KPAD * HDIM, 1.0f, 0, 0, stream);
    {
      size_t n = (size_t)TROWS * DMODEL;
      reshape_attn_kernel<<<CDIV(n, 256), 256, 0, stream>>>(attn, obuf16, KPAD, n);
      gemm(obuf16, Wo, bo, cbuf, TROWS, DMODEL, DMODEL, 1, 0, 0, 0, 1.0f, 0, 0, stream);
      residual_add_kernel<<<CDIV(n, 256), 256, 0, stream>>>(x, cbuf, n);
      layernorm_kernel<<<TROWSP, 256, 0, stream>>>(x, g2, b2, hbuf16, nullptr, DMODEL, TROWS);
      gemm(hbuf16, Wf, bf, mlph16, TROWS, FFDIM, DMODEL, 1, 0, 0, 0, 1.0f, 1, 1, stream);
      gemm(mlph16, Wc, bc, cbuf, TROWS, DMODEL, FFDIM, 1, 0, 0, 0, 1.0f, 0, 0, stream);
      residual_add_kernel<<<CDIV(n, 256), 256, 0, stream>>>(x, cbuf, n);
    }
  }

  // ---- final: ln_post(gv) @ proj, reorder to [B, NMASK, 512] ----
  layernorm_kernel<<<GVROWS, 256, 0, stream>>>(gv, lnp_g, lnp_b, hbuf16, nullptr, DMODEL, GVROWS);
  gemm(hbuf16, wp16, nullptr, fcbuf, GVROWS, OUTD, DMODEL, 1, 0, 0, 0, 1.0f, 0, 0, stream);
  final_reorder_kernel<<<CDIV((size_t)BATCH * NMASK * OUTD, 256), 256, 0, stream>>>(fcbuf, out);
}
